// MyLSTM_69002944578236
// MI455X (gfx1250) — matrix operand order repaired and checked by reading, not compiled, not run on hardware
//
#include <hip/hip_runtime.h>

// Problem dims (match reference)
#define Bsz 128
#define Tsz 512
#define Fsz 512
#define Hsz 1024
#define CT  16   // timesteps per x-GEMM precompute chunk

typedef __attribute__((ext_vector_type(16))) __bf16        v16bf;
typedef __attribute__((ext_vector_type(8)))  float          v8f;
typedef __attribute__((ext_vector_type(8)))  unsigned short v8u;
typedef __attribute__((ext_vector_type(16))) unsigned short v16u;

// ---------------------------------------------------------------- helpers
static __device__ __forceinline__ unsigned short f2bf(float f) {
    // native f32 -> bf16 (RNE) -- lowers to v_cvt_pk_bf16_f32
    __bf16 b = (__bf16)f;
    return __builtin_bit_cast(unsigned short, b);
}

static __device__ __forceinline__ float fast_sigmoid(float x) {
    return 1.0f / (1.0f + __expf(-x));
}

static __device__ __forceinline__ float fast_tanh(float x) {
    float ax = fabsf(x);
    float e  = __expf(-2.0f * ax);
    float t  = (1.0f - e) / (1.0f + e);
    return copysignf(t, x);
}

// A-tile (16x32 bf16, MxK) loader from bf16 memory, per CDNA5 ISA 7.12.2:
// lanes 0-15 (row m): K = k0+0..7 (elems 0-7), k0+16..23 (elems 8-15)
// lanes 16-31        : K = k0+8..15,            k0+24..31
static __device__ __forceinline__ v16bf load_a_bf16(const unsigned short* __restrict__ base,
                                                    long ld, int row, int k0, int lane) {
    int m = lane & 15, hs = lane >> 4;
    const unsigned short* p = base + (long)(row + m) * ld + (k0 + hs * 8);
    v8u lo = *(const v8u*)(p);
    v8u hi = *(const v8u*)(p + 16);
    v16u a;
#pragma unroll
    for (int i = 0; i < 8; ++i) { a[i] = lo[i]; a[8 + i] = hi[i]; }
    return __builtin_bit_cast(v16bf, a);
}

// Same A tile but sourced from f32 memory, cvt in-register (RNE).
static __device__ __forceinline__ v16bf load_a_f32(const float* __restrict__ base,
                                                   long ld, int row, int k0, int lane) {
    int m = lane & 15, hs = lane >> 4;
    const float* p = base + (long)(row + m) * ld + (k0 + hs * 8);
    float4 f0 = *(const float4*)(p);
    float4 f1 = *(const float4*)(p + 4);
    float4 f2 = *(const float4*)(p + 16);
    float4 f3 = *(const float4*)(p + 20);
    v16u a;
    a[0]  = f2bf(f0.x); a[1]  = f2bf(f0.y); a[2]  = f2bf(f0.z); a[3]  = f2bf(f0.w);
    a[4]  = f2bf(f1.x); a[5]  = f2bf(f1.y); a[6]  = f2bf(f1.z); a[7]  = f2bf(f1.w);
    a[8]  = f2bf(f2.x); a[9]  = f2bf(f2.y); a[10] = f2bf(f2.z); a[11] = f2bf(f2.w);
    a[12] = f2bf(f3.x); a[13] = f2bf(f3.y); a[14] = f2bf(f3.z); a[15] = f2bf(f3.w);
    return __builtin_bit_cast(v16bf, a);
}

// B-tile (32x16 bf16, KxN) from torch-layout W[Nrows, K] row-major:
// lane n (0-15) holds column n, K = k0+0..7, k0+16..23; lanes 16-31 hold K = k0+8..15, k0+24..31.
// Column n of B == row (nrow+n) of W -> two 16B loads per lane, 32B apart.
static __device__ __forceinline__ v16bf load_b(const unsigned short* __restrict__ W,
                                               int K, int nrow, int k0, int lane) {
    int n = lane & 15, kg = lane >> 4;
    const unsigned short* p = W + (long)(nrow + n) * K + (k0 + kg * 8);
    v8u lo = *(const v8u*)(p);
    v8u hi = *(const v8u*)(p + 16);
    v16u b;
#pragma unroll
    for (int i = 0; i < 8; ++i) { b[i] = lo[i]; b[8 + i] = hi[i]; }
    return __builtin_bit_cast(v16bf, b);
}

#define WMMA_BF16(a, b, c) \
    __builtin_amdgcn_wmma_f32_16x16x32_bf16(false, (a), false, (b), (short)0, (c), false, false)

// Software-pipelined GEMM loop: acc[g] += A(16xK) @ W[g*H+nt .. +16, 0..K]^T.
// 2-deep ping-pong (phase A loop-carried, phase B iteration-local): the next
// K-step's 10 b128 loads fly while the current 4 chained WMMAs execute, and
// there is no register rotation in source, so the accumulator chain maps
// in-place. Requires K/32 even and >= 2 (here K is 512 or 1024).
template <bool AF32>
static __device__ __forceinline__ void gemm4(v8f (&acc)[4], const void* __restrict__ Abase,
                                             long lda, int mt,
                                             const unsigned short* __restrict__ W,
                                             int K, int nt, int lane) {
    v16bf aA  = AF32 ? load_a_f32((const float*)Abase, lda, mt, 0, lane)
                     : load_a_bf16((const unsigned short*)Abase, lda, mt, 0, lane);
    v16bf bA0 = load_b(W, K, 0 * Hsz + nt, 0, lane);
    v16bf bA1 = load_b(W, K, 1 * Hsz + nt, 0, lane);
    v16bf bA2 = load_b(W, K, 2 * Hsz + nt, 0, lane);
    v16bf bA3 = load_b(W, K, 3 * Hsz + nt, 0, lane);

    int k0 = 32;
    for (; k0 < K - 32; k0 += 64) {
        // phase B tiles for step k0 (local to the iteration)
        v16bf aB  = AF32 ? load_a_f32((const float*)Abase, lda, mt, k0, lane)
                         : load_a_bf16((const unsigned short*)Abase, lda, mt, k0, lane);
        v16bf bB0 = load_b(W, K, 0 * Hsz + nt, k0, lane);
        v16bf bB1 = load_b(W, K, 1 * Hsz + nt, k0, lane);
        v16bf bB2 = load_b(W, K, 2 * Hsz + nt, k0, lane);
        v16bf bB3 = load_b(W, K, 3 * Hsz + nt, k0, lane);
        acc[0] = WMMA_BF16(aA, bA0, acc[0]);
        acc[1] = WMMA_BF16(aA, bA1, acc[1]);
        acc[2] = WMMA_BF16(aA, bA2, acc[2]);
        acc[3] = WMMA_BF16(aA, bA3, acc[3]);
        // reload phase A in place for step k0+32
        aA  = AF32 ? load_a_f32((const float*)Abase, lda, mt, k0 + 32, lane)
                   : load_a_bf16((const unsigned short*)Abase, lda, mt, k0 + 32, lane);
        bA0 = load_b(W, K, 0 * Hsz + nt, k0 + 32, lane);
        bA1 = load_b(W, K, 1 * Hsz + nt, k0 + 32, lane);
        bA2 = load_b(W, K, 2 * Hsz + nt, k0 + 32, lane);
        bA3 = load_b(W, K, 3 * Hsz + nt, k0 + 32, lane);
        acc[0] = WMMA_BF16(aB, bB0, acc[0]);
        acc[1] = WMMA_BF16(aB, bB1, acc[1]);
        acc[2] = WMMA_BF16(aB, bB2, acc[2]);
        acc[3] = WMMA_BF16(aB, bB3, acc[3]);
    }
    // tail: exactly one step (K-32) left beyond the resident phase-A tiles
    {
        v16bf aB  = AF32 ? load_a_f32((const float*)Abase, lda, mt, k0, lane)
                         : load_a_bf16((const unsigned short*)Abase, lda, mt, k0, lane);
        v16bf bB0 = load_b(W, K, 0 * Hsz + nt, k0, lane);
        v16bf bB1 = load_b(W, K, 1 * Hsz + nt, k0, lane);
        v16bf bB2 = load_b(W, K, 2 * Hsz + nt, k0, lane);
        v16bf bB3 = load_b(W, K, 3 * Hsz + nt, k0, lane);
        acc[0] = WMMA_BF16(aA, bA0, acc[0]);
        acc[1] = WMMA_BF16(aA, bA1, acc[1]);
        acc[2] = WMMA_BF16(aA, bA2, acc[2]);
        acc[3] = WMMA_BF16(aA, bA3, acc[3]);
        acc[0] = WMMA_BF16(aB, bB0, acc[0]);
        acc[1] = WMMA_BF16(aB, bB1, acc[1]);
        acc[2] = WMMA_BF16(aB, bB2, acc[2]);
        acc[3] = WMMA_BF16(aB, bB3, acc[3]);
    }
}

// ---------------------------------------------------------------- kernels

// One LSTM layer, one timestep.
//   A1MODE 0: input GEMM from f32 A1 (layer-1 x_t), bias init
//   A1MODE 1: input GEMM from bf16 A1 (layer-2 h1), bias init
//   A1MODE 2: no input GEMM; accumulators seeded from tiled precomputed f32
//             partial gx = x_t @ W_ih^T + b_ih + b_hh  (layer-1 fast path)
// Each wave owns one 16x16 (batch x hidden) tile for ALL FOUR gates, so the
// LSTM cell update happens entirely in registers.
template <int A1MODE, bool WRITE_HF>
__global__ __launch_bounds__(128)
void lstm_layer_k(const float* __restrict__ A1f,
                  const unsigned short* __restrict__ A1b,
                  const float* __restrict__ gx,            // tiled partial (mode 2)
                  int K1, long lda1,
                  const unsigned short* __restrict__ A2b,  // recurrent h (bf16), ld = Hsz
                  const unsigned short* __restrict__ Wih,  // [4H, K1] bf16 (modes 0/1)
                  const unsigned short* __restrict__ Whh,  // [4H, H]  bf16
                  const float* __restrict__ b_ih, const float* __restrict__ b_hh,
                  float* __restrict__ c_state,             // [B, H] f32 (in/out)
                  unsigned short* __restrict__ h_out_bf,   // [B, H] bf16 (out)
                  float* __restrict__ h_out_f)             // [B, H] f32 (out)
{
    const int lane = threadIdx.x & 31;
    const int wave = threadIdx.x >> 5;
    const int wid  = blockIdx.x * 4 + wave;    // 0..511
    const int mt   = (wid >> 6) << 4;          // batch-row tile base (0..112)
    const int nt   = (wid & 63) << 4;          // within-H column tile base (0..1008)

    const int col   = nt + (lane & 15);
    const int rbase = ((lane >> 4) << 3);      // 0 or 8 (C/D row group per lane half)

    v8f acc[4];
    if constexpr (A1MODE == 2) {
        // wave-tile-major layout: tile = (mt/16)*64 + nt/16, 4KB per tile,
        // lane-private 128B block -> 8 x b128 loads
        const float* p = gx + ((long)((mt >> 4) * 64 + (nt >> 4)) << 10) + lane * 32;
#pragma unroll
        for (int g = 0; g < 4; ++g) acc[g] = *(const v8f*)(p + g * 8);
    } else {
#pragma unroll
        for (int g = 0; g < 4; ++g) {
            float bv = b_ih[g * Hsz + col] + b_hh[g * Hsz + col];
#pragma unroll
            for (int r = 0; r < 8; ++r) acc[g][r] = bv;
        }
    }

    // ---- input GEMM: A1 @ Wih^T (skipped in mode 2)
    if constexpr (A1MODE == 0) {
        gemm4<true>(acc, A1f, lda1, mt, Wih, K1, nt, lane);
    } else if constexpr (A1MODE == 1) {
        gemm4<false>(acc, A1b, lda1, mt, Wih, K1, nt, lane);
    }

    // ---- recurrent GEMM: h_prev @ Whh^T
    gemm4<false>(acc, A2b, (long)Hsz, mt, Whh, Hsz, nt, lane);

    // ---- fused LSTM cell: i,f,g,o for the same (m, col) are lane-aligned
#pragma unroll
    for (int r = 0; r < 8; ++r) {
        int  m   = mt + rbase + r;
        long idx = (long)m * Hsz + col;
        float iv = fast_sigmoid(acc[0][r]);
        float fv = fast_sigmoid(acc[1][r]);
        float gv = fast_tanh(acc[2][r]);
        float ov = fast_sigmoid(acc[3][r]);
        float c  = fv * c_state[idx] + iv * gv;
        c_state[idx] = c;
        float h = ov * fast_tanh(c);
        h_out_bf[idx] = f2bf(h);
        if constexpr (WRITE_HF) h_out_f[idx] = h;
    }
}

// Parallel precompute of the dependency-free x-GEMM for CT timesteps:
//   partial(tloc, b, g*H+n) = x[b, t0+tloc, :] @ W_ih1[g*H+n, :] + b_ih + b_hh
// stored in wave-tile-major layout (4KB per 16x16x4-gate tile, lane-private
// 128B blocks) so both producer stores and consumer loads are b128.
// 8192 waves (vs 512 in the scan kernels) -> runs near the WMMA roofline.
__global__ __launch_bounds__(128)
void gates_x_precompute(const float* __restrict__ x, int t0,
                        const unsigned short* __restrict__ Wih,
                        const float* __restrict__ b_ih, const float* __restrict__ b_hh,
                        float* __restrict__ gx)
{
    const int lane = threadIdx.x & 31;
    const int wave = threadIdx.x >> 5;
    const int wid  = blockIdx.x * 4 + wave;    // 0..8191
    const int mtg  = (wid >> 6) << 4;          // row tile over CT*B = 2048 rows
    const int nt   = (wid & 63) << 4;

    const int tloc = mtg >> 7;                 // which timestep in the chunk
    const int b0   = mtg & 127;                // batch-row tile base
    const float* Abase = x + (long)(t0 + tloc) * Fsz;  // row b at Abase + b*T*F

    const int col = nt + (lane & 15);

    v8f acc[4];
#pragma unroll
    for (int g = 0; g < 4; ++g) {
        float bv = b_ih[g * Hsz + col] + b_hh[g * Hsz + col];
#pragma unroll
        for (int r = 0; r < 8; ++r) acc[g][r] = bv;
    }

    gemm4<true>(acc, Abase, (long)Tsz * Fsz, b0, Wih, Fsz, nt, lane);

    const int tile = (tloc * (Bsz / 16) + (b0 >> 4)) * 64 + (nt >> 4);
    float* p = gx + ((long)tile << 10) + lane * 32;
#pragma unroll
    for (int g = 0; g < 4; ++g) *(v8f*)(p + g * 8) = acc[g];
}

__global__ __launch_bounds__(256)
void cvt_f32_bf16(const float* __restrict__ in, unsigned short* __restrict__ out, int n) {
    int i = blockIdx.x * 256 + threadIdx.x;
    if (i < n) out[i] = f2bf(in[i]);
}

__global__ __launch_bounds__(256)
void final_linear(const float* __restrict__ h2, const float* __restrict__ Wl,
                  const float* __restrict__ bl, float* __restrict__ out) {
    __shared__ float red[256];
    int b = blockIdx.x;
    float s = 0.0f;
    for (int k = threadIdx.x; k < Hsz; k += 256)
        s += h2[(long)b * Hsz + k] * Wl[k];
    red[threadIdx.x] = s;
    __syncthreads();
    for (int off = 128; off > 0; off >>= 1) {
        if (threadIdx.x < off) red[threadIdx.x] += red[threadIdx.x + off];
        __syncthreads();
    }
    if (threadIdx.x == 0) out[b] = red[0] + bl[0];
}

// ---------------------------------------------------------------- launch
extern "C" void kernel_launch(void* const* d_in, const int* in_sizes, int n_in,
                              void* d_out, int out_size, void* d_ws, size_t ws_size,
                              hipStream_t stream) {
    (void)in_sizes; (void)n_in; (void)out_size;

    const float* x     = (const float*)d_in[0];
    const float* W_ih1 = (const float*)d_in[1];
    const float* W_hh1 = (const float*)d_in[2];
    const float* b_ih1 = (const float*)d_in[3];
    const float* b_hh1 = (const float*)d_in[4];
    const float* W_ih2 = (const float*)d_in[5];
    const float* W_hh2 = (const float*)d_in[6];
    const float* b_ih2 = (const float*)d_in[7];
    const float* b_hh2 = (const float*)d_in[8];
    const float* W_lin = (const float*)d_in[9];
    const float* b_lin = (const float*)d_in[10];
    float* out = (float*)d_out;

    // workspace carve-out (256B aligned offsets)
    size_t off = 0;
    char* ws = (char*)d_ws;
    auto alc = [&](size_t bytes) -> void* {
        void* p = ws + off;
        off += (bytes + 255) & ~(size_t)255;
        return p;
    };
    unsigned short* Wih1b = (unsigned short*)alc((size_t)4 * Hsz * Fsz * 2);
    unsigned short* Whh1b = (unsigned short*)alc((size_t)4 * Hsz * Hsz * 2);
    unsigned short* Wih2b = (unsigned short*)alc((size_t)4 * Hsz * Hsz * 2);
    unsigned short* Whh2b = (unsigned short*)alc((size_t)4 * Hsz * Hsz * 2);
    unsigned short* h1b[2] = {(unsigned short*)alc((size_t)Bsz * Hsz * 2),
                              (unsigned short*)alc((size_t)Bsz * Hsz * 2)};
    unsigned short* h2b[2] = {(unsigned short*)alc((size_t)Bsz * Hsz * 2),
                              (unsigned short*)alc((size_t)Bsz * Hsz * 2)};
    float* c1  = (float*)alc((size_t)Bsz * Hsz * 4);
    float* c2  = (float*)alc((size_t)Bsz * Hsz * 4);
    float* h2f = (float*)alc((size_t)Bsz * Hsz * 4);
    // chunked x-gates buffer last, so the fast path is used only if ws allows
    float* gx  = (float*)alc((size_t)CT * Bsz * 4 * Hsz * 4);   // 32 MB
    const bool use_pre = (off <= ws_size);   // deterministic tier choice

    // weights -> bf16 (L2-resident for the whole scan: ~28 MB vs 192 MB L2)
    {
        int n1 = 4 * Hsz * Fsz;
        int n2 = 4 * Hsz * Hsz;
        cvt_f32_bf16<<<(n1 + 255) / 256, 256, 0, stream>>>(W_ih1, Wih1b, n1);
        cvt_f32_bf16<<<(n2 + 255) / 256, 256, 0, stream>>>(W_hh1, Whh1b, n2);
        cvt_f32_bf16<<<(n2 + 255) / 256, 256, 0, stream>>>(W_ih2, Wih2b, n2);
        cvt_f32_bf16<<<(n2 + 255) / 256, 256, 0, stream>>>(W_hh2, Whh2b, n2);
    }

    // zero initial states (deterministic per call)
    hipMemsetAsync(h1b[0], 0, (size_t)Bsz * Hsz * 2, stream);
    hipMemsetAsync(h1b[1], 0, (size_t)Bsz * Hsz * 2, stream);
    hipMemsetAsync(h2b[0], 0, (size_t)Bsz * Hsz * 2, stream);
    hipMemsetAsync(h2b[1], 0, (size_t)Bsz * Hsz * 2, stream);
    hipMemsetAsync(c1, 0, (size_t)Bsz * Hsz * 4, stream);
    hipMemsetAsync(c2, 0, (size_t)Bsz * Hsz * 4, stream);

    if (use_pre) {
        // hoist the dependency-free x-GEMM out of the scan, 16 timesteps at a
        // time; stream order serializes producer/consumer so one buffer is safe
        const long gx_t_stride = (long)(Bsz / 16) * 64 * 1024;  // floats per timestep
        for (int c = 0; c < Tsz / CT; ++c) {
            gates_x_precompute<<<2048, 128, 0, stream>>>(x, c * CT, Wih1b,
                                                         b_ih1, b_hh1, gx);
            for (int tl = 0; tl < CT; ++tl) {
                int t = c * CT + tl;
                int rp = t & 1, wp = rp ^ 1;
                lstm_layer_k<2, false><<<128, 128, 0, stream>>>(
                    nullptr, nullptr, gx + tl * gx_t_stride, 0, 0,
                    h1b[rp], nullptr, Whh1b, nullptr, nullptr,
                    c1, h1b[wp], nullptr);
                lstm_layer_k<1, true><<<128, 128, 0, stream>>>(
                    nullptr, h1b[wp], nullptr, Hsz, (long)Hsz,
                    h2b[rp], Wih2b, Whh2b, b_ih2, b_hh2,
                    c2, h2b[wp], h2f);
            }
        }
    } else {
        for (int t = 0; t < Tsz; ++t) {
            int rp = t & 1, wp = rp ^ 1;
            lstm_layer_k<0, false><<<128, 128, 0, stream>>>(
                x + (long)t * Fsz, nullptr, nullptr, Fsz, (long)Tsz * Fsz,
                h1b[rp], Wih1b, Whh1b, b_ih1, b_hh1,
                c1, h1b[wp], nullptr);
            lstm_layer_k<1, true><<<128, 128, 0, stream>>>(
                nullptr, h1b[wp], nullptr, Hsz, (long)Hsz,
                h2b[rp], Wih2b, Whh2b, b_ih2, b_hh2,
                c2, h2b[wp], h2f);
        }
    }

    // linear on last timestep's h2
    final_linear<<<Bsz, 256, 0, stream>>>(h2f, W_lin, b_lin, out);
}
